// ShawAttention_32847909880334
// MI455X (gfx1250) — matrix operand order repaired and checked by reading, not compiled, not run on hardware
//
#include <hip/hip_runtime.h>
#include <hip/hip_bf16.h>

// ---------------------------------------------------------------------------
// Shaw relative attention on gfx1250 (MI455X).
// All GEMMs via v_wmma_f32_16x16x32_f16 (wave32); whole-tile LDS fills via the
// Tensor Data Mover (tensor_load_to_lds + s_wait_tensorcnt) where available;
// streaming GEMM K-tiles prefetched with global_prefetch.
// B=512, L=64, D=1024, H=16, dk=64.
// ---------------------------------------------------------------------------

typedef __attribute__((ext_vector_type(16))) _Float16 v16h;
typedef __attribute__((ext_vector_type(8)))  float    v8f;
typedef __attribute__((ext_vector_type(4)))  unsigned v4u;
typedef __attribute__((ext_vector_type(8)))  int      v8i;
typedef __attribute__((ext_vector_type(4)))  int      v4i;

#define NB 512          // batch
#define NL 64           // seq len
#define ND 1024         // model dim
#define NH 16           // heads
#define DK 64           // head dim
#define NBH (NB*NH)     // 8192

// ---- Tensor Data Mover availability -----------------------------------------
#if defined(__has_builtin)
#if __has_builtin(__builtin_amdgcn_tensor_load_to_lds)
#define USE_TDM 1
#endif
#endif
#ifndef USE_TDM
#define USE_TDM 0
#endif

static __device__ __forceinline__ void wait_tensorcnt0() {
#if defined(__has_builtin) && __has_builtin(__builtin_amdgcn_s_wait_tensorcnt)
    __builtin_amdgcn_s_wait_tensorcnt(0);
#else
    asm volatile("s_wait_tensorcnt 0x0" ::: "memory");
#endif
}

#if USE_TDM
// DMA a tile of `rows` rows x rowUnits (8-byte units) from global (row stride
// gStrideUnits, 8B units) into LDS at `lds`, inserting 4 DWORDs of pad after
// every 32 DWORDs (=> f16 row stride 72 halves, matching our LDS tiles).
// Descriptor packing per CDNA5 ISA 8.3/8.4 (D# group0/group1), groups 2/3 NULL.
static __device__ __forceinline__ void tdm_load_tile(const _Float16* g, const _Float16* lds,
                                                     unsigned rows, unsigned rowUnits,
                                                     unsigned gStrideUnits) {
    unsigned long long ga = (unsigned long long)(size_t)g;
    unsigned ldsOff = (unsigned)(size_t)lds;   // low 32 bits of generic LDS ptr = LDS offset
    v4u g0;
    g0.x = 1u;                                           // count=1, user descriptor
    g0.y = ldsOff;                                       // lds_addr
    g0.z = (unsigned)ga;                                 // global_addr[31:0]
    g0.w = (unsigned)((ga >> 32) & 0x01FFFFFFu) | (2u << 30);  // global_addr[56:32] | type=2
    v8i g1;
    g1[0] = (int)((3u << 16)      // data_size = 8 bytes
                | (1u << 20)      // pad_enable
                | (4u << 22)      // pad_interval: 32 DWORDs (one 128B row)
                | (3u << 25));    // pad_amount: 4 DWORDs (16B)
    g1[1] = (int)(rowUnits << 16);        // [63:48] tensor_dim0[15:0]
    g1[2] = (int)(rows << 16);            // [79:64] tdim0 hi=0 | [95:80] tensor_dim1[15:0]
    g1[3] = (int)(rowUnits << 16);        // [111:96] tdim1 hi=0 | [127:112] tile_dim0
    g1[4] = (int)rows;                    // [143:128] tile_dim1 | [159:144] tile_dim2=0
    g1[5] = (int)gStrideUnits;            // [191:160] tensor_dim0_stride[31:0]
    g1[6] = 0;                            // stride hi | tensor_dim1_stride lo
    g1[7] = 0;
    v4i z4 = {0, 0, 0, 0};
#if __clang_major__ >= 23
    v8i z8 = {0, 0, 0, 0, 0, 0, 0, 0};
    __builtin_amdgcn_tensor_load_to_lds(g0, g1, z4, z4, z8, 0);
#else
    __builtin_amdgcn_tensor_load_to_lds(g0, g1, z4, z4, 0);
#endif
}
#endif

static __device__ __forceinline__ v8f wmma16(v16h a, v16h b, v8f c) {
    // D = A(16x32 f16) * B(32x16 f16) + C(16x16 f32)
    return __builtin_amdgcn_wmma_f32_16x16x32_f16(
        false, a, false, b, (short)0, c, false, false);
}

// global(f16, row-major) -> LDS tile copy, 16B chunks, 256 threads
static __device__ __forceinline__ void g2s(const _Float16* __restrict__ g, int gstride,
                                           _Float16* __restrict__ s, int sstride,
                                           int rows, int cols, int tid) {
    const int chunks = rows * cols / 8;
    for (int c = tid; c < chunks; c += 256) {
        int lin = c * 8;
        int r   = lin / cols;
        int col = lin - r * cols;
        *(int4*)(s + r * sstride + col) = *(const int4*)(g + r * gstride + col);
    }
}

// A fragment (16x32 f16): lane holds row M=lane&15; halves 0..7 = K[g*8..g*8+7],
// halves 8..15 = K[16+g*8..16+g*8+7], g = lane>>4  (ISA 7.12.2 table)
static __device__ __forceinline__ v16h load_a(const _Float16* __restrict__ s, int stride,
                                              int m0, int k0, int lane) {
    int m = lane & 15;
    int g = (lane >> 4) & 1;
    const _Float16* p0 = s + (m0 + m) * stride + k0 + g * 8;
    const _Float16* p1 = p0 + 16;
    v16h a;
#pragma unroll
    for (int t = 0; t < 8; ++t) { a[t] = p0[t]; a[8 + t] = p1[t]; }
    return a;
}

// B fragment (32x16 f16): lane holds column N=lane&15; halves 0..7 = K[g*8..g*8+7],
// halves 8..15 = K[16+g*8..16+g*8+7], g = lane>>4
static __device__ __forceinline__ v16h load_b(const _Float16* __restrict__ s, int stride,
                                              int k0, int n0, int lane) {
    int n = lane & 15;
    int g = (lane >> 4) & 1;
    const _Float16* p = s + (k0 + g * 8) * stride + n0 + n;
    v16h b;
#pragma unroll
    for (int t = 0; t < 8; ++t) { b[t] = p[t * stride]; b[8 + t] = p[(16 + t) * stride]; }
    return b;
}

// ---------------------------------------------------------------------------
// K0a: f32 -> f16 convert (grid-stride)
// ---------------------------------------------------------------------------
__global__ __launch_bounds__(256) void k_convert(const float* __restrict__ src,
                                                 _Float16* __restrict__ dst, int n) {
    for (int i = blockIdx.x * 256 + threadIdx.x; i < n; i += gridDim.x * 256)
        dst[i] = (_Float16)src[i];
}

// ---------------------------------------------------------------------------
// K0b: gather rel tables.  rKt[i][d][j] = rel_k[idx[i,j]][d]  (B-operand layout
// for the rel-score GEMM);  rVn[i][j][d] = rel_v[idx[i,j]][d].
// ---------------------------------------------------------------------------
__global__ __launch_bounds__(256) void k_gather_rel(const int* __restrict__ idx,
                                                    const float* __restrict__ relk,
                                                    const float* __restrict__ relv,
                                                    _Float16* __restrict__ rKt,
                                                    _Float16* __restrict__ rVn) {
    const int i    = blockIdx.x;          // 0..63
    const int j    = threadIdx.x >> 2;    // 0..63
    const int part = threadIdx.x & 3;     // 0..3
    const int id   = idx[i * NL + j];
#pragma unroll
    for (int t = 0; t < 16; ++t) {
        int d = part * 16 + t;
        rKt[i * (DK * NL) + d * NL + j] = (_Float16)relk[id * DK + d];  // [i][d][j]
        rVn[i * (NL * DK) + j * DK + d] = (_Float16)relv[id * DK + d];  // [i][j][d]
    }
}

// ---------------------------------------------------------------------------
// K1: fused QKV projection.  grid = (B, H, 3).  One WG computes a 64x64 tile
// of x@W (M = 64 tokens of batch b, N = head h's 64 dims, K = 1024), then
// RMSNorms q/k rows and stores:
//   which=0: Qn[bh][i][d]   (normalized, *q_scale)
//   which=1: Kd[bh][d][i]   (normalized, *k_scale, TRANSPOSED -> B operand)
//   which=2: Vn[bh][i][d]
// ---------------------------------------------------------------------------
__global__ __launch_bounds__(256) void k_qkv(const _Float16* __restrict__ xh,
                                             const _Float16* __restrict__ Wq,
                                             const _Float16* __restrict__ Wk,
                                             const _Float16* __restrict__ Wv,
                                             const float* __restrict__ qscale,
                                             const float* __restrict__ kscale,
                                             _Float16* __restrict__ Qn,
                                             _Float16* __restrict__ Kd,
                                             _Float16* __restrict__ Vn) {
    const int b = blockIdx.x, h = blockIdx.y, which = blockIdx.z;
    const _Float16* W = (which == 0) ? Wq : (which == 1) ? Wk : Wv;

    __shared__ alignas(16) _Float16 As[64 * 40];
    __shared__ alignas(16) _Float16 Bs[32 * 72];
    __shared__ float Cs[64 * 66];
    __shared__ float red[64 * 4];

    const int tid  = threadIdx.x;
    const int lane = tid & 31;
    const int wave = tid >> 5;
    const int m0 = (wave & 3) * 16;
    const int n0 = (wave >> 2) * 32;

    v8f c0 = {}, c1 = {};
    const _Float16* Abase = xh + (size_t)(b * NL) * ND;
    const _Float16* Bbase = W + h * DK;

    for (int kt = 0; kt < ND; kt += 32) {
        __syncthreads();
        g2s(Abase + kt, ND, As, 40, 64, 32, tid);
        g2s(Bbase + (size_t)kt * ND, ND, Bs, 72, 32, 64, tid);
        if (kt + 32 < ND) {   // prefetch next K-tile (global_prefetch)
            __builtin_prefetch(Abase + (kt + 32) + (tid >> 2) * ND + (tid & 3) * 8, 0, 3);
            __builtin_prefetch(Bbase + (size_t)(kt + 32 + (tid >> 3)) * ND + (tid & 7) * 8, 0, 3);
        }
        __syncthreads();
        v16h a  = load_a(As, 40, m0, 0, lane);
        v16h b0 = load_b(Bs, 72, 0, n0, lane);
        v16h b1 = load_b(Bs, 72, 0, n0 + 16, lane);
        c0 = wmma16(a, b0, c0);
        c1 = wmma16(a, b1, c1);
    }
    __syncthreads();

    // spill C tiles to LDS (C layout: M = v + 8*(lane>>4), N = lane&15)
    {
        int nn = lane & 15, mb = (lane >> 4) * 8;
#pragma unroll
        for (int v = 0; v < 8; ++v) {
            int m = m0 + mb + v;
            Cs[m * 66 + n0 + nn]      = c0[v];
            Cs[m * 66 + n0 + 16 + nn] = c1[v];
        }
    }
    __syncthreads();

    const size_t bh = (size_t)(b * NH + h);
    if (which == 2) {           // V: plain f16 store, natural layout
        for (int e = tid; e < 64 * 64; e += 256) {
            int r = e >> 6, cidx = e & 63;
            Vn[bh * (NL * DK) + r * DK + cidx] = (_Float16)Cs[r * 66 + cidx];
        }
    } else {                    // Q/K: RMSNorm over dk then store
        int r = tid & 63, qp = tid >> 6;
        float s = 0.f;
#pragma unroll
        for (int t = 0; t < 16; ++t) {
            float x = Cs[r * 66 + qp * 16 + t];
            s += x * x;
        }
        red[r * 4 + qp] = s;
        __syncthreads();
        float tot = red[r * 4] + red[r * 4 + 1] + red[r * 4 + 2] + red[r * 4 + 3];
        float inv = rsqrtf(tot * (1.0f / DK) + 1e-6f);
        const float* sc = (which == 0) ? qscale : kscale;
#pragma unroll
        for (int t = 0; t < 16; ++t) {
            int d = qp * 16 + t;
            float val = Cs[r * 66 + d] * inv * sc[d];
            if (which == 0) Qn[bh * (NL * DK) + r * DK + d] = (_Float16)val;
            else            Kd[bh * (NL * DK) + d * NL + r] = (_Float16)val;
        }
    }
}

// ---------------------------------------------------------------------------
// K2: rel scores.  For fixed i: REL[bh, j] = Q_i[bh, d] @ rKt[i][d][j].
// grid = (NBH/64, NL).  M=64 bh rows (global row stride 4096 halves), K=N=64.
// ---------------------------------------------------------------------------
__global__ __launch_bounds__(256) void k_relscore(const _Float16* __restrict__ Qn,
                                                  const _Float16* __restrict__ rKt,
                                                  _Float16* __restrict__ rel) {
    const int bh0 = blockIdx.x * 64;
    const int i   = blockIdx.y;
    __shared__ alignas(16) _Float16 As[64 * 72];
    __shared__ alignas(16) _Float16 Bs[64 * 72];
    const int tid = threadIdx.x, lane = tid & 31, wave = tid >> 5;
    const int m0 = (wave & 3) * 16, n0 = (wave >> 2) * 32;

#if USE_TDM
    if (wave == 0) {   // one DMA issue per tile (TENSORcnt-tracked)
        tdm_load_tile(Qn + (size_t)bh0 * (NL * DK) + i * DK, As, 64, 16, 1024);
        tdm_load_tile(rKt + (size_t)i * (DK * NL),           Bs, 64, 16, 16);
        wait_tensorcnt0();
    }
#else
    g2s(Qn + (size_t)bh0 * (NL * DK) + i * DK, NL * DK, As, 72, 64, 64, tid);
    g2s(rKt + (size_t)i * (DK * NL), NL, Bs, 72, 64, 64, tid);
#endif
    __syncthreads();

    v8f c0 = {}, c1 = {};
#pragma unroll
    for (int kt = 0; kt < 64; kt += 32) {
        v16h a  = load_a(As, 72, m0, kt, lane);
        v16h b0 = load_b(Bs, 72, kt, n0, lane);
        v16h b1 = load_b(Bs, 72, kt, n0 + 16, lane);
        c0 = wmma16(a, b0, c0);
        c1 = wmma16(a, b1, c1);
    }
    int nn = lane & 15, mb = (lane >> 4) * 8;
#pragma unroll
    for (int v = 0; v < 8; ++v) {
        size_t gm = (size_t)(bh0 + m0 + mb + v);
        rel[gm * (NL * NL) + i * NL + n0 + nn]      = (_Float16)c0[v];
        rel[gm * (NL * NL) + i * NL + n0 + 16 + nn] = (_Float16)c1[v];
    }
}

// ---------------------------------------------------------------------------
// K3: attention core per (b,h): S = Q@K^T + rel, *1/8, softmax rows,
// O1 = P@V.  Writes probs (f16) and partial O1 (f16, overwrites rel slice).
// ---------------------------------------------------------------------------
__global__ __launch_bounds__(256) void k_attn(const _Float16* __restrict__ Qn,
                                              const _Float16* __restrict__ Kd,
                                              const _Float16* __restrict__ Vn,
                                              const _Float16* __restrict__ rel,
                                              _Float16* __restrict__ probs,
                                              _Float16* __restrict__ att1) {
    const size_t bh = blockIdx.x;
    __shared__ alignas(16) _Float16 Qs[64 * 72];
    __shared__ alignas(16) _Float16 Ks[64 * 72];   // [d][j]  (pre-transposed K)
    __shared__ alignas(16) _Float16 Vs[64 * 72];   // [j][d]
    __shared__ alignas(16) _Float16 Ps[64 * 72];   // probs [i][j]
    __shared__ float Ss[64 * 66];

    const int tid = threadIdx.x, lane = tid & 31, wave = tid >> 5;
    const int m0 = (wave & 3) * 16, n0 = (wave >> 2) * 32;

#if USE_TDM
    if (wave == 0) {
        tdm_load_tile(Qn + bh * (NL * DK), Qs, 64, 16, 16);
        tdm_load_tile(Kd + bh * (NL * DK), Ks, 64, 16, 16);
        tdm_load_tile(Vn + bh * (NL * DK), Vs, 64, 16, 16);
        wait_tensorcnt0();
    }
#else
    g2s(Qn + bh * (NL * DK), DK, Qs, 72, 64, 64, tid);
    g2s(Kd + bh * (NL * DK), NL, Ks, 72, 64, 64, tid);
    g2s(Vn + bh * (NL * DK), DK, Vs, 72, 64, 64, tid);
#endif
    __syncthreads();

    // ---- S = Q @ K^T ----
    v8f c0 = {}, c1 = {};
#pragma unroll
    for (int kt = 0; kt < DK; kt += 32) {
        v16h a  = load_a(Qs, 72, m0, kt, lane);
        v16h b0 = load_b(Ks, 72, kt, n0, lane);
        v16h b1 = load_b(Ks, 72, kt, n0 + 16, lane);
        c0 = wmma16(a, b0, c0);
        c1 = wmma16(a, b1, c1);
    }
    {   // add rel bias, apply 1/sqrt(dk), spill to LDS
        int nn = lane & 15, mb = (lane >> 4) * 8;
        const _Float16* rbase = rel + bh * (NL * NL);
#pragma unroll
        for (int v = 0; v < 8; ++v) {
            int m = m0 + mb + v;
            float r0 = (float)rbase[m * NL + n0 + nn];
            float r1 = (float)rbase[m * NL + n0 + 16 + nn];
            Ss[m * 66 + n0 + nn]      = (c0[v] + r0) * 0.125f;
            Ss[m * 66 + n0 + 16 + nn] = (c1[v] + r1) * 0.125f;
        }
    }
    __syncthreads();

    // ---- row softmax (one thread per row) ----
    if (tid < 64) {
        int r = tid;
        float mx = -3.0e38f;
        for (int j = 0; j < NL; ++j) mx = fmaxf(mx, Ss[r * 66 + j]);
        float sum = 0.f;
        for (int j = 0; j < NL; ++j) {
            float e = __expf(Ss[r * 66 + j] - mx);
            Ss[r * 66 + j] = e;
            sum += e;
        }
        float inv = 1.0f / sum;
        _Float16* pg = probs + bh * (NL * NL) + r * NL;
        for (int j = 0; j < NL; ++j) {
            _Float16 p = (_Float16)(Ss[r * 66 + j] * inv);
            Ps[r * 72 + j] = p;
            pg[j]          = p;          // for the rel_v GEMM (K4)
        }
    }
    __syncthreads();

    // ---- O1 = P @ V ----
    v8f o0 = {}, o1 = {};
#pragma unroll
    for (int kt = 0; kt < NL; kt += 32) {
        v16h a  = load_a(Ps, 72, m0, kt, lane);
        v16h b0 = load_b(Vs, 72, kt, n0, lane);
        v16h b1 = load_b(Vs, 72, kt, n0 + 16, lane);
        o0 = wmma16(a, b0, o0);
        o1 = wmma16(a, b1, o1);
    }
    {
        int nn = lane & 15, mb = (lane >> 4) * 8;
        _Float16* ab = att1 + bh * (NL * DK);
#pragma unroll
        for (int v = 0; v < 8; ++v) {
            int m = m0 + mb + v;
            ab[m * DK + n0 + nn]      = (_Float16)o0[v];
            ab[m * DK + n0 + 16 + nn] = (_Float16)o1[v];
        }
    }
}

// ---------------------------------------------------------------------------
// K4: rel_v term.  For fixed i: O2[bh, d] = P_i[bh, j] @ rVn[i][j][d];
// out = O1 + O2, written to ah[(b*64+i)*1024 + h*64 + d] (f16, pre-Wo).
// ---------------------------------------------------------------------------
__global__ __launch_bounds__(256) void k_relv(const _Float16* __restrict__ probs,
                                              const _Float16* __restrict__ rVn,
                                              const _Float16* __restrict__ att1,
                                              _Float16* __restrict__ ah) {
    const int bh0 = blockIdx.x * 64;
    const int i   = blockIdx.y;
    __shared__ alignas(16) _Float16 As[64 * 72];
    __shared__ alignas(16) _Float16 Bs[64 * 72];
    const int tid = threadIdx.x, lane = tid & 31, wave = tid >> 5;
    const int m0 = (wave & 3) * 16, n0 = (wave >> 2) * 32;

#if USE_TDM
    if (wave == 0) {
        tdm_load_tile(probs + (size_t)bh0 * (NL * NL) + i * NL, As, 64, 16, 1024);
        tdm_load_tile(rVn + (size_t)i * (NL * DK),              Bs, 64, 16, 16);
        wait_tensorcnt0();
    }
#else
    g2s(probs + (size_t)bh0 * (NL * NL) + i * NL, NL * NL, As, 72, 64, 64, tid);
    g2s(rVn + (size_t)i * (NL * DK), DK, Bs, 72, 64, 64, tid);
#endif
    __syncthreads();

    v8f c0 = {}, c1 = {};
#pragma unroll
    for (int kt = 0; kt < 64; kt += 32) {
        v16h a  = load_a(As, 72, m0, kt, lane);
        v16h b0 = load_b(Bs, 72, kt, n0, lane);
        v16h b1 = load_b(Bs, 72, kt, n0 + 16, lane);
        c0 = wmma16(a, b0, c0);
        c1 = wmma16(a, b1, c1);
    }
    int nn = lane & 15, mb = (lane >> 4) * 8;
#pragma unroll
    for (int v = 0; v < 8; ++v) {
        int gm = bh0 + m0 + mb + v;          // global bh index
        int b = gm >> 4, h = gm & 15;
        float s0 = c0[v] + (float)att1[(size_t)gm * (NL * DK) + i * DK + n0 + nn];
        float s1 = c1[v] + (float)att1[(size_t)gm * (NL * DK) + i * DK + n0 + 16 + nn];
        size_t row = (size_t)(b * NL + i) * ND + h * DK;
        ah[row + n0 + nn]      = (_Float16)s0;
        ah[row + n0 + 16 + nn] = (_Float16)s1;
    }
}

// ---------------------------------------------------------------------------
// K5: output projection.  OUT[32768, 1024] = ah @ Wo, f32 result.
// grid = (512, 16); each WG a 64x64 tile, K = 1024.
// ---------------------------------------------------------------------------
__global__ __launch_bounds__(256) void k_outproj(const _Float16* __restrict__ ah,
                                                 const _Float16* __restrict__ Wo,
                                                 float* __restrict__ out) {
    const int r0 = blockIdx.x * 64;
    const int c0c = blockIdx.y * 64;
    __shared__ alignas(16) _Float16 As[64 * 40];
    __shared__ alignas(16) _Float16 Bs[32 * 72];
    const int tid = threadIdx.x, lane = tid & 31, wave = tid >> 5;
    const int m0 = (wave & 3) * 16, n0 = (wave >> 2) * 32;

    const _Float16* Abase = ah + (size_t)r0 * ND;
    const _Float16* Bbase = Wo + c0c;

    v8f c0 = {}, c1 = {};
    for (int kt = 0; kt < ND; kt += 32) {
        __syncthreads();
        g2s(Abase + kt, ND, As, 40, 64, 32, tid);
        g2s(Bbase + (size_t)kt * ND, ND, Bs, 72, 32, 64, tid);
        if (kt + 32 < ND) {   // prefetch next K-tile
            __builtin_prefetch(Abase + (kt + 32) + (tid >> 2) * ND + (tid & 3) * 8, 0, 3);
            __builtin_prefetch(Bbase + (size_t)(kt + 32 + (tid >> 3)) * ND + (tid & 7) * 8, 0, 3);
        }
        __syncthreads();
        v16h a  = load_a(As, 40, m0, 0, lane);
        v16h b0 = load_b(Bs, 72, 0, n0, lane);
        v16h b1 = load_b(Bs, 72, 0, n0 + 16, lane);
        c0 = wmma16(a, b0, c0);
        c1 = wmma16(a, b1, c1);
    }
    int nn = lane & 15, mb = (lane >> 4) * 8;
#pragma unroll
    for (int v = 0; v < 8; ++v) {
        size_t m = (size_t)(r0 + m0 + mb + v);
        out[m * ND + c0c + n0 + nn]      = c0[v];
        out[m * ND + c0c + n0 + 16 + nn] = c1[v];
    }
}

// ---------------------------------------------------------------------------
extern "C" void kernel_launch(void* const* d_in, const int* in_sizes, int n_in,
                              void* d_out, int out_size, void* d_ws, size_t ws_size,
                              hipStream_t stream) {
    const float* x       = (const float*)d_in[0];
    const int*   relIdx  = (const int*)d_in[1];
    const float* Wq      = (const float*)d_in[2];
    const float* Wk      = (const float*)d_in[3];
    const float* Wv      = (const float*)d_in[4];
    const float* Wo      = (const float*)d_in[5];
    const float* qscale  = (const float*)d_in[6];
    const float* kscale  = (const float*)d_in[7];
    const float* relk    = (const float*)d_in[8];
    const float* relv    = (const float*)d_in[9];
    float* out = (float*)d_out;

    char* ws = (char*)d_ws;
    const size_t szAct = (size_t)NB * NL * ND * 2;   // 67 MB  (f16 activations)
    const size_t szW   = (size_t)ND * ND * 2;        // 2 MB
    const size_t szRel = (size_t)NL * NL * DK * 2;   // 512 KB

    size_t off = 0;
    _Float16* xh  = (_Float16*)(ws + off); off += szAct;  // aliased later by probs
    _Float16* Wqh = (_Float16*)(ws + off); off += szW;
    _Float16* Wkh = (_Float16*)(ws + off); off += szW;
    _Float16* Wvh = (_Float16*)(ws + off); off += szW;
    _Float16* Woh = (_Float16*)(ws + off); off += szW;
    _Float16* Qn  = (_Float16*)(ws + off); off += szAct;
    _Float16* Kd  = (_Float16*)(ws + off); off += szAct;
    _Float16* Vn  = (_Float16*)(ws + off); off += szAct;
    _Float16* rKt = (_Float16*)(ws + off); off += szRel;
    _Float16* rVn = (_Float16*)(ws + off); off += szRel;
    _Float16* rel = (_Float16*)(ws + off); off += szAct; // [bh][i][j]; reused as att1
    _Float16* ah  = (_Float16*)(ws + off); off += szAct; // pre-Wo activations

    _Float16* probs = xh;   // xh dead after K1
    _Float16* att1  = rel;  // rel consumed inside K3 before att1 write (same WG)

    // K0: conversions + rel gather
    k_convert<<<dim3(4096), dim3(256), 0, stream>>>(x,  xh,  NB * NL * ND);
    k_convert<<<dim3(1024), dim3(256), 0, stream>>>(Wq, Wqh, ND * ND);
    k_convert<<<dim3(1024), dim3(256), 0, stream>>>(Wk, Wkh, ND * ND);
    k_convert<<<dim3(1024), dim3(256), 0, stream>>>(Wv, Wvh, ND * ND);
    k_convert<<<dim3(1024), dim3(256), 0, stream>>>(Wo, Woh, ND * ND);
    k_gather_rel<<<dim3(NL), dim3(256), 0, stream>>>(relIdx, relk, relv, rKt, rVn);

    // K1: fused QKV projection + RMSNorm (+K transpose)
    k_qkv<<<dim3(NB, NH, 3), dim3(256), 0, stream>>>(xh, Wqh, Wkh, Wvh,
                                                     qscale, kscale, Qn, Kd, Vn);
    // K2: rel score GEMMs (batched over bh, one i per block.y)
    k_relscore<<<dim3(NBH / 64, NL), dim3(256), 0, stream>>>(Qn, rKt, rel);
    // K3: attention core per (b,h)
    k_attn<<<dim3(NBH), dim3(256), 0, stream>>>(Qn, Kd, Vn, rel, probs, att1);
    // K4: rel_v GEMMs + combine
    k_relv<<<dim3(NBH / 64, NL), dim3(256), 0, stream>>>(probs, rVn, att1, ah);
    // K5: output projection (f32 out)
    k_outproj<<<dim3((NB * NL) / 64, ND / 64), dim3(256), 0, stream>>>(ah, Woh, out);
}
